// KroneckerMessage_76871324663920
// MI455X (gfx1250) — compile-verified, matrix operand order checked by reading, not run on hardware
//
#include <hip/hip_runtime.h>

typedef __attribute__((ext_vector_type(16))) _Float16 v16h;
typedef __attribute__((ext_vector_type(8)))  float    v8f;
typedef __attribute__((ext_vector_type(2)))  _Float16 h2;

#define LN_EPS 1e-5f
#define NCHUNK 15            // K' = 20*24 = 480 = 15 chunks of 32
#define NFRAG  (NCHUNK * 8)  // 120 B fragments
#define ROW_STRIDE 24        // halfs per h row (20 + zero pad)

// ---------------------------------------------------------------- kernel 0
__global__ void zero_kernel(float* __restrict__ p, int n) {
    int i = blockIdx.x * blockDim.x + threadIdx.x;
    if (i < n) p[i] = 0.0f;
}

// ---------------------------------------------------------------- kernel 1
// h[r, 0:20] = relu(LN(node_feat[r] @ W_node + b_node)) as f16, rows padded to 24
__global__ void __launch_bounds__(256) node_proj_kernel(
    const float* __restrict__ x, const float* __restrict__ Wn,
    const float* __restrict__ bn, const float* __restrict__ gn,
    const float* __restrict__ betan, _Float16* __restrict__ h, int N)
{
    __shared__ float Wl[256 * 20];
    for (int i = threadIdx.x; i < 256 * 20; i += 256) Wl[i] = Wn[i];
    __syncthreads();

    int r = blockIdx.x * 256 + threadIdx.x;
    if (r >= N) return;

    float acc[20];
    #pragma unroll
    for (int j = 0; j < 20; ++j) acc[j] = bn[j];

    const float4* xr = (const float4*)(x + (size_t)r * 256);
    for (int kb = 0; kb < 64; ++kb) {
        float4 xv = xr[kb];
        const float* w0 = &Wl[(kb * 4 + 0) * 20];
        const float* w1 = &Wl[(kb * 4 + 1) * 20];
        const float* w2 = &Wl[(kb * 4 + 2) * 20];
        const float* w3 = &Wl[(kb * 4 + 3) * 20];
        #pragma unroll
        for (int j = 0; j < 20; ++j)
            acc[j] += xv.x * w0[j] + xv.y * w1[j] + xv.z * w2[j] + xv.w * w3[j];
    }

    float s1 = 0.f, s2 = 0.f;
    #pragma unroll
    for (int j = 0; j < 20; ++j) { s1 += acc[j]; s2 += acc[j] * acc[j]; }
    float mu  = s1 * (1.0f / 20.0f);
    float var = s2 * (1.0f / 20.0f) - mu * mu;
    float rs  = rsqrtf(var + LN_EPS);

    _Float16* hr = h + (size_t)r * ROW_STRIDE;
    #pragma unroll
    for (int j = 0; j < 20; ++j) {
        float y = (acc[j] - mu) * rs * gn[j] + betan[j];
        hr[j] = (_Float16)fmaxf(y, 0.0f);
    }
    hr[20] = (_Float16)0.f; hr[21] = (_Float16)0.f;
    hr[22] = (_Float16)0.f; hr[23] = (_Float16)0.f;
}

// ---------------------------------------------------------------- kernel 2
// Pre-swizzle W_kron [400,128] f32 into per-lane v16h B fragments for
// v_wmma_f32_16x16x32_f16 under the padded K ordering k' = a*24 + b
// (b in 0..23, b >= 20 -> 0). Fragment f = kk*8 + nt ; lane L:
// n = nt*16 + (L&15), half = L>>4 ;
// vgpr pair v: k' = kk*32 + (v>=4 ? 16 : 0) + (v&3)*2 + half*8 (+0,+1)
__global__ void swizzle_wkron_kernel(const float* __restrict__ Wk,
                                     _Float16* __restrict__ swb)
{
    int gid = blockIdx.x * 256 + threadIdx.x;          // (f, L) pair id
    if (gid >= NFRAG * 32) return;
    int f = gid >> 5, L = gid & 31;
    int kk = f >> 3, nt = f & 7;
    int n = nt * 16 + (L & 15);
    int half = L >> 4;
    _Float16* o = swb + (size_t)gid * 16;
    #pragma unroll
    for (int v = 0; v < 8; ++v) {
        int kp = kk * 32 + ((v >= 4) ? 16 : 0) + (v & 3) * 2 + half * 8;
        #pragma unroll
        for (int s = 0; s < 2; ++s) {
            int a = (kp + s) / 24;
            int b = (kp + s) - a * 24;
            o[v * 2 + s] = (b < 20) ? (_Float16)Wk[(a * 20 + b) * 128 + n]
                                    : (_Float16)0.f;
        }
    }
}

// ---------------------------------------------------------------- kernel 3
// One 8-element run of the A fragment under k' = a*24 + b:
// runs start at multiples of 8, 8 | 24 -> single a per run, contiguous b
// window (b0 in {0,8,16}) -> one 16B-aligned ds_load_b128 + one ds_load_u16.
__device__ __forceinline__ void build_run(v16h& A, int base, int k0,
                                          const _Float16* hs,
                                          const _Float16* hd)
{
    unsigned a  = (unsigned)k0 / 24u;
    int      b0 = k0 - (int)a * 24;
    _Float16 s  = hs[a];
    uint4 q = *(const uint4*)(hd + b0);
    h2 sv; sv.x = s; sv.y = s;
    h2 r0 = sv * __builtin_bit_cast(h2, q.x);
    h2 r1 = sv * __builtin_bit_cast(h2, q.y);
    h2 r2 = sv * __builtin_bit_cast(h2, q.z);
    h2 r3 = sv * __builtin_bit_cast(h2, q.w);
    A[base + 0] = r0.x; A[base + 1] = r0.y;
    A[base + 2] = r1.x; A[base + 3] = r1.y;
    A[base + 4] = r2.x; A[base + 5] = r2.y;
    A[base + 6] = r3.x; A[base + 7] = r3.y;
}

__global__ void __launch_bounds__(256) edge_kernel(
    const _Float16* __restrict__ h, const _Float16* __restrict__ swb,
    const float* __restrict__ bk, const float* __restrict__ gk,
    const float* __restrict__ bek,
    const int* __restrict__ src, const int* __restrict__ dst,
    float* __restrict__ out, int E, int nTiles, int totalWaves)
{
    __shared__ __align__(16) _Float16 Bs[NFRAG * 512];          // 122880 B
    __shared__ __align__(16) _Float16 HS[8][16 * ROW_STRIDE];   // 6144 B
    __shared__ __align__(16) _Float16 HD[8][16 * ROW_STRIDE];   // 6144 B
    __shared__ int DI[8][16];

    const int tid = threadIdx.x;

    // block-wide: stage swizzled B into LDS (16B vector copies, L2-resident src)
    {
        const uint4* g = (const uint4*)swb;
        uint4* l = (uint4*)Bs;
        for (int i = tid; i < NFRAG * 512 * 2 / 16; i += 256) l[i] = g[i];
    }
    __syncthreads();

    const int wave = tid >> 5;
    const int lane = tid & 31;
    const int m    = lane & 15;
    const int half = lane >> 4;

    _Float16* hsw = HS[wave];
    _Float16* hdw = HD[wave];
    int*      diw = DI[wave];

    // per-lane column params (col = nt*16 + m), broadcast within half-waves
    float gl[8], bl[8], kl[8];
    #pragma unroll
    for (int nt = 0; nt < 8; ++nt) {
        int col = nt * 16 + m;
        gl[nt] = gk[col]; bl[nt] = bek[col]; kl[nt] = bk[col];
    }

    for (int t = blockIdx.x * 8 + wave; t < nTiles; t += totalWaves) {
        // ---- stage 16 src rows (lanes 0-15) and 16 dst rows (lanes 16-31)
        int e = t * 16 + m;
        int eok = e < E;
        int idx;
        if (lane < 16) idx = eok ? src[e] : 0;
        else          { idx = eok ? dst[e] : 0; diw[m] = eok ? idx : -1; }

        const uint4* hrp = (const uint4*)(h + (size_t)idx * ROW_STRIDE);
        uint4 p0 = hrp[0], p1 = hrp[1], p2 = hrp[2];  // halfs 0..23 (20..23 zero)
        if (!eok) { p0 = make_uint4(0,0,0,0); p1 = p0; p2 = p0; }

        unsigned* dr = (lane < 16) ? (unsigned*)(hsw + m * ROW_STRIDE)
                                   : (unsigned*)(hdw + m * ROW_STRIDE);
        dr[0] = p0.x; dr[1] = p0.y; dr[2] = p0.z; dr[3] = p0.w;
        dr[4] = p1.x; dr[5] = p1.y; dr[6] = p1.z; dr[7] = p1.w;
        dr[8] = p2.x; dr[9] = p2.y; dr[10] = p2.z; dr[11] = p2.w;

        // Opaque per-iteration values: block LICM of (a) the 240 B-fragment
        // ds_loads, (b) the per-run a/b0 address computations. Recomputing a
        // few VALU ops per tile is far cheaper than holding them live (which
        // previously spilled to scratch / packed masks into readlane chains).
        unsigned bofs = lane * 16;                    // halfs
        unsigned hofs = m * ROW_STRIDE;               // halfs
        int halfo = half;
        asm volatile("" : "+v"(bofs), "+v"(hofs), "+v"(halfo));
        const _Float16* Bsl   = Bs + bofs;
        const _Float16* hsrow = hsw + hofs;
        const _Float16* hdrow = hdw + hofs;

        // ---- K loop: 15 chunks of 32, 8 N-tiles -> 120 WMMA per tile
        v8f acc[8];
        #pragma unroll
        for (int nt = 0; nt < 8; ++nt) acc[nt] = (v8f)(0.0f);

        #pragma unroll
        for (int kk = 0; kk < NCHUNK; ++kk) {
            v16h A;
            build_run(A, 0, kk * 32 +      halfo * 8, hsrow, hdrow);
            build_run(A, 8, kk * 32 + 16 + halfo * 8, hsrow, hdrow);
            #pragma unroll
            for (int nt = 0; nt < 8; ++nt) {
                const v16h* bp = (const v16h*)(Bsl + (kk * 8 + nt) * 512);
                acc[nt] = __builtin_amdgcn_wmma_f32_16x16x32_f16(
                    false, A, false, *bp, (short)0, acc[nt], false, false);
            }
        }

        // ---- + b_kron, then LayerNorm over 128 cols per edge row
        float s1[8], s2[8];
        #pragma unroll
        for (int r = 0; r < 8; ++r) { s1[r] = 0.f; s2[r] = 0.f; }
        #pragma unroll
        for (int nt = 0; nt < 8; ++nt) {
            #pragma unroll
            for (int r = 0; r < 8; ++r) {
                float v = acc[nt][r] + kl[nt];
                acc[nt][r] = v;
                s1[r] += v; s2[r] += v * v;
            }
        }
        #pragma unroll
        for (int o = 1; o < 16; o <<= 1) {
            #pragma unroll
            for (int r = 0; r < 8; ++r) {
                s1[r] += __shfl_xor(s1[r], o, 32);
                s2[r] += __shfl_xor(s2[r], o, 32);
            }
        }

        int dstr[8];
        #pragma unroll
        for (int r = 0; r < 8; ++r) dstr[r] = diw[half * 8 + r];  // LDS broadcast

        #pragma unroll
        for (int r = 0; r < 8; ++r) {
            float mu  = s1[r] * (1.0f / 128.0f);
            float var = s2[r] * (1.0f / 128.0f) - mu * mu;
            float rs  = rsqrtf(var + LN_EPS);
            int d = dstr[r];
            #pragma unroll
            for (int nt = 0; nt < 8; ++nt) {
                float y = (acc[nt][r] - mu) * rs * gl[nt] + bl[nt];
                y = fmaxf(y, 0.0f);
                if (d >= 0)
                    atomicAdd(out + (size_t)d * 128 + nt * 16 + m, y);
            }
        }
    }
}

// ---------------------------------------------------------------- launch
extern "C" void kernel_launch(void* const* d_in, const int* in_sizes, int n_in,
                              void* d_out, int out_size, void* d_ws, size_t ws_size,
                              hipStream_t stream)
{
    const float* node_feat = (const float*)d_in[0];
    const float* W_node    = (const float*)d_in[1];
    const float* b_node    = (const float*)d_in[2];
    const float* g_node    = (const float*)d_in[3];
    const float* beta_node = (const float*)d_in[4];
    const float* W_kron    = (const float*)d_in[5];
    const float* b_kron    = (const float*)d_in[6];
    const float* g_kron    = (const float*)d_in[7];
    const float* beta_kron = (const float*)d_in[8];
    const int*   src       = (const int*)d_in[9];
    const int*   dst       = (const int*)d_in[10];

    const int N = in_sizes[0] / 256;
    const int E = in_sizes[10];
    float* out = (float*)d_out;

    // workspace: h (N x 24 halfs) | swizzled W_kron fragments
    _Float16* h = (_Float16*)d_ws;
    size_t hBytes = (((size_t)N * ROW_STRIDE * 2) + 255) & ~(size_t)255;
    _Float16* swb = (_Float16*)((char*)d_ws + hBytes);

    // 0) zero output (atomic accumulation target)
    zero_kernel<<<(out_size + 255) / 256, 256, 0, stream>>>(out, out_size);

    // 1) node projection -> h (f16)
    node_proj_kernel<<<(N + 255) / 256, 256, 0, stream>>>(
        node_feat, W_node, b_node, g_node, beta_node, h, N);

    // 2) swizzle W_kron into WMMA B-fragment layout (padded K' = 480)
    swizzle_wkron_kernel<<<(NFRAG * 32 + 255) / 256, 256, 0, stream>>>(W_kron, swb);

    // 3) fused kron-GEMM + LN + ReLU + atomic scatter
    int nTiles = (E + 15) / 16;
    int blocks = (nTiles + 7) / 8;
    if (blocks > 1024) blocks = 1024;
    if (blocks < 1) blocks = 1;
    edge_kernel<<<blocks, 256, 0, stream>>>(
        h, swb, b_kron, g_kron, beta_kron, src, dst, out, E, nTiles, blocks * 8);
}
